// AdaptiveHistogramGuidedAttention_84061099917693
// MI455X (gfx1250) — matrix operand order repaired and checked by reading, not compiled, not run on hardware
//
#include <hip/hip_runtime.h>
#include <hip/hip_bf16.h>

typedef __attribute__((ext_vector_type(16))) _Float16     v16h;
typedef __attribute__((ext_vector_type(8)))  float        v8f;
typedef __attribute__((ext_vector_type(4)))  unsigned int v4u;
typedef __attribute__((ext_vector_type(8)))  int          v8i_t;
typedef __attribute__((ext_vector_type(4)))  int          v4i_t;

#define B_     8
#define C_     64
#define HALF_  32
#define H_     256
#define W_     256
#define N_     65536
#define OC3_   192
#define HEADS_ 8
#define D_     8
#define EPS_   1e-12f

__device__ __forceinline__ float sigm_(float x) { return 1.f / (1.f + __expf(-x)); }
__device__ __forceinline__ float silu_(float x) { return x * sigm_(x); }

// A-matrix (16x32 f16) lane->K mapping: contiguous, k = 16*hf + e (matches B).
__device__ __forceinline__ int a_ksrc(int e, int hf) {
    int v = e >> 1;
    (void)v;
    return hf * 16 + e;
}

// Stage a 16x64 f32 weight tile into LDS in WMMA-fragment order:
// lAf[m][ks*32 + half*16 + e] so each lane fragment is one contiguous 32B run.
__device__ __forceinline__ void stage_a_frag(const float* __restrict__ wsrc, int m0,
                                             _Float16 (*lAf)[64], int tid) {
    for (int i = tid; i < 16 * 64; i += 256) {
        int m = i >> 6, q = i & 63;
        int ks = q >> 5, rem = q & 31;
        int hf = rem >> 4, e = rem & 15;
        lAf[m][q] = (_Float16)wsrc[(size_t)(m0 + m) * C_ + ks * 32 + a_ksrc(e, hf)];
    }
}

// TDM: issue a 2-D tensor_load_to_lds for a rows x cols f32 tile (row stride = srow elems).
// amdgpu-toolchain (clang-23) arity: (v4u g0, v8i g1, v4i g2, v4i g3, v8i pad, i32 cpol)
__device__ __forceinline__ void tdm_load_tile_f32(const float* gsrc, unsigned lds_byte_addr,
                                                  unsigned cols, unsigned rows,
                                                  unsigned tensor_d0, unsigned tensor_d1,
                                                  unsigned long long srow) {
    unsigned long long ga = (unsigned long long)(size_t)gsrc;
    v4u g0;
    g0[0] = 1u;                                             // count=1, user mode
    g0[1] = lds_byte_addr;                                  // LDS dest (bytes)
    g0[2] = (unsigned)(ga & 0xffffffffu);                   // global_addr[31:0]
    g0[3] = (unsigned)((ga >> 32) & 0x01ffffffu) | (2u << 30);  // addr[56:32] | type=2
    v8i_t g1;
    g1[0] = (int)(2u << 16);                                // data_size = 4 bytes
    g1[1] = (int)((tensor_d0 & 0xffffu) << 16);             // tensor_dim0[15:0]
    g1[2] = (int)(((tensor_d0 >> 16) & 0xffffu) | ((tensor_d1 & 0xffffu) << 16));
    g1[3] = (int)(((tensor_d1 >> 16) & 0xffffu) | (cols << 16));   // tile_dim0
    g1[4] = (int)rows;                                      // tile_dim1 (tile_dim2=0)
    g1[5] = (int)(srow & 0xffffffffu);                      // dim0 stride lo
    g1[6] = (int)((srow >> 32) & 0xffffu);                  // dim0 stride hi
    g1[7] = 0;
    v4i_t z4 = {0, 0, 0, 0};
    v8i_t z8 = {0, 0, 0, 0, 0, 0, 0, 0};
    __builtin_amdgcn_tensor_load_to_lds(g0, g1, z4, z4, z8, 0);
}

// ---------------------------------------------------------------- copy x -> xw
__global__ __launch_bounds__(256) void k_copy(const float* __restrict__ src,
                                              float* __restrict__ dst) {
    size_t gid = (size_t)blockIdx.x * 256 + threadIdx.x;
    dst[gid] = src[gid];
}

__global__ __launch_bounds__(256) void k_zero(float* __restrict__ p, int n) {
    int i = blockIdx.x * 256 + threadIdx.x;
    if (i < n) p[i] = 0.f;
}

// ------------------------------------------------- 256-element LDS bitonic sort
__device__ __forceinline__ void bitonic256(float* sv, int* si) {
    const int t = threadIdx.x;
    for (int k = 2; k <= 256; k <<= 1) {
        for (int j = k >> 1; j > 0; j >>= 1) {
            __syncthreads();
            int l = t ^ j;
            if (l > t) {
                float v0 = sv[t], v1 = sv[l];
                int   i0 = si[t], i1 = si[l];
                bool asc = ((t & k) == 0);
                bool gt  = (v0 > v1) || (v0 == v1 && i0 > i1);
                if (gt == asc) { sv[t] = v1; sv[l] = v0; si[t] = i1; si[l] = i0; }
            }
        }
    }
    __syncthreads();
}

__global__ __launch_bounds__(256) void k_sort_h(float* __restrict__ xw,
                                                int* __restrict__ idx_h) {
    __shared__ float sv[256];
    __shared__ int   si[256];
    const int blk = blockIdx.x;           // b*HALF*W + ch*W + w
    const int w   = blk & 255;
    const int bc  = blk >> 8;
    const int b   = bc >> 5, ch = bc & 31;
    const int t   = threadIdx.x;
    const size_t base = ((size_t)(b * C_ + ch) * H_) * W_ + w;
    sv[t] = xw[base + (size_t)t * W_];
    si[t] = t;
    __syncthreads();
    bitonic256(sv, si);
    xw[base + (size_t)t * W_] = sv[t];
    idx_h[((size_t)bc * H_ + t) * W_ + w] = si[t];
}

__global__ __launch_bounds__(256) void k_sort_w(float* __restrict__ xw,
                                                int* __restrict__ idx_w) {
    __shared__ float sv[256];
    __shared__ int   si[256];
    const int blk = blockIdx.x;           // b*HALF*H + ch*H + h
    const int h   = blk & 255;
    const int bc  = blk >> 8;
    const int b   = bc >> 5, ch = bc & 31;
    const int t   = threadIdx.x;
    const size_t base = ((size_t)(b * C_ + ch) * H_ + h) * W_;
    sv[t] = xw[base + t];
    si[t] = t;
    __syncthreads();
    bitonic256(sv, si);
    xw[base + t] = sv[t];
    idx_w[((size_t)bc * H_ + h) * W_ + t] = si[t];
}

// ---------------- qkv GEMM: (192x64)x(64xN), B tile via TDM, fragment-order LDS
__global__ __launch_bounds__(256) void k_qkv_gemm(const float* __restrict__ xw,
                                                  const float* __restrict__ w_qkv,
                                                  float* __restrict__ qkv) {
    __shared__ float    lBx[64][128];       // raw f32 tile landed by TDM
    __shared__ _Float16 lBt[128][64];       // transposed fragment-order f16 tile
    __shared__ _Float16 lAf[16][64];        // fragment-order A tile
    const int b  = blockIdx.z;
    const int m0 = blockIdx.y * 16;
    const int n0 = blockIdx.x * 128;
    const int tid = threadIdx.x;

    if (tid < 32) {   // one wave issues the DMA; TDM ignores EXEC
        tdm_load_tile_f32(xw + (size_t)b * C_ * N_ + n0,
                          (unsigned)(size_t)(&lBx[0][0]),
                          128u, 64u, (unsigned)N_, (unsigned)C_,
                          (unsigned long long)N_);
        __builtin_amdgcn_s_wait_tensorcnt((short)0);
    }
    stage_a_frag(w_qkv, m0, lAf, tid);
    __syncthreads();
    // convert + transpose into fragment order (each lane frag = contiguous 32B)
    for (int i = tid; i < 64 * 128; i += 256) {
        int c = i >> 7, j = i & 127;
        lBt[j][c] = (_Float16)lBx[c][j];
    }
    __syncthreads();

    const int wave = tid >> 5, lane = tid & 31;
    const int nn = lane & 15, half = lane >> 4;
    const int m = lane & 15;
    const int col_l = wave * 16 + nn;
    v8f acc = {};
#pragma unroll
    for (int ks = 0; ks < 2; ++ks) {
        v16h a  = *(const v16h*)&lAf[m][ks * 32 + half * 16];
        v16h bm = *(const v16h*)&lBt[col_l][ks * 32 + half * 16];
        acc = __builtin_amdgcn_wmma_f32_16x16x32_f16(false, a, false, bm,
                                                     (short)0, acc, false, false);
    }
    const int col   = n0 + col_l;
    const int mbase = m0 + half * 8;
#pragma unroll
    for (int r = 0; r < 8; ++r)
        qkv[((size_t)b * OC3_ + mbase + r) * N_ + col] = acc[r];
}

// ---------------------------------------------------------- depthwise 3x3 conv
__global__ __launch_bounds__(256) void k_dwconv(const float* __restrict__ qkv,
                                                const float* __restrict__ w_dw,
                                                float* __restrict__ qkvd) {
    size_t gid = (size_t)blockIdx.x * 256 + threadIdx.x;   // B*OC3*N
    int    p   = (int)(gid & (N_ - 1));
    size_t row = gid >> 16;                                 // b*OC3 + o
    int    o   = (int)(row % OC3_);
    int    h   = p >> 8, w = p & 255;
    const float* src = qkv + (row << 16);
    float acc = 0.f;
#pragma unroll
    for (int ky = 0; ky < 3; ++ky) {
        int hy = h + ky - 1;
        if ((unsigned)hy >= (unsigned)H_) continue;
#pragma unroll
        for (int kx = 0; kx < 3; ++kx) {
            int wx = w + kx - 1;
            if ((unsigned)wx >= (unsigned)W_) continue;
            acc += src[hy * W_ + wx] * w_dw[o * 9 + ky * 3 + kx];
        }
    }
    qkvd[gid] = acc;
}

// ------------------------------------ v sort: init keys/idx, global bitonic pass
__global__ __launch_bounds__(256) void k_vinit(const float* __restrict__ qkvd,
                                               float* __restrict__ vkey,
                                               int* __restrict__ vidx) {
    size_t gid = (size_t)blockIdx.x * 256 + threadIdx.x;   // B*C*N
    int p = (int)(gid & (N_ - 1));
    int row = (int)(gid >> 16);
    int b = row >> 6, c = row & 63;
    vkey[gid] = qkvd[((size_t)b * OC3_ + 128 + c) * N_ + p];
    vidx[gid] = p;
}

__global__ __launch_bounds__(256) void k_bitonic_pass(float* __restrict__ key,
                                                      int* __restrict__ idx,
                                                      int k, int j) {
    size_t gid = (size_t)blockIdx.x * 256 + threadIdx.x;   // B*C*N
    int    i   = (int)(gid & (N_ - 1));
    size_t row = gid >> 16;
    int l = i ^ j;
    if (l > i) {
        size_t base = row << 16;
        float v0 = key[base + i], v1 = key[base + l];
        int   i0 = idx[base + i], i1 = idx[base + l];
        bool asc = ((i & k) == 0);
        bool gt  = (v0 > v1) || (v0 == v1 && i0 > i1);
        if (gt == asc) {
            key[base + i] = v1; key[base + l] = v0;
            idx[base + i] = i1; idx[base + l] = i0;
        }
    }
}

__global__ __launch_bounds__(256) void k_gather_qk(const float* __restrict__ qkvd,
                                                   const int* __restrict__ vidx,
                                                   float* __restrict__ q_s,
                                                   float* __restrict__ k_s) {
    size_t gid = (size_t)blockIdx.x * 256 + threadIdx.x;   // B*C*N
    int row = (int)(gid >> 16);
    int b = row >> 6, c = row & 63;
    int src = vidx[gid];
    q_s[gid] = qkvd[((size_t)b * OC3_ + c) * N_ + src];
    k_s[gid] = qkvd[((size_t)b * OC3_ + 64 + c) * N_ + src];
}

// -------------------------------------------------- per-row inverse l2 norms
__global__ __launch_bounds__(256) void k_norms(const float* __restrict__ q_s,
                                               const float* __restrict__ k_s,
                                               float* __restrict__ nrm) {
    __shared__ float red[256];
    const int which = blockIdx.x >> 9;   // 0: q, 1: k
    const int row   = blockIdx.x & 511;
    const float* src = (which == 0 ? q_s : k_s) + (size_t)row * N_;
    float s = 0.f;
    for (int n = threadIdx.x; n < N_; n += 256) { float v = src[n]; s += v * v; }
    red[threadIdx.x] = s;
    __syncthreads();
    for (int off = 128; off; off >>= 1) {
        if (threadIdx.x < off) red[threadIdx.x] += red[threadIdx.x + off];
        __syncthreads();
    }
    if (threadIdx.x == 0)
        nrm[which * 512 + row] = 1.f / fmaxf(sqrtf(red[0]), EPS_);
}

// ---- attention logits: K=65536 split over 32 blocks/(b,head); atomic partials
__global__ __launch_bounds__(32) void k_attn_partial(const float* __restrict__ q_s,
                                                     const float* __restrict__ k_s,
                                                     float* __restrict__ attn) {
    const int bh    = blockIdx.y;            // 0..63
    const int chunk = blockIdx.x;            // 0..31
    const int b = bh >> 3, head = bh & 7;
    const int lane = threadIdx.x;
    const int m = lane & 15, half = lane >> 4;
    const float* qrow = q_s + ((size_t)(b * C_ + head * D_ + (m & 7))) * N_;
    const float* krow = k_s + ((size_t)(b * C_ + head * D_ + (m & 7))) * N_;
    const bool mok = m < 8;
    v8f acc = {};
    const int nbeg = chunk * (N_ / 32), nend = nbeg + (N_ / 32);
    for (int n0 = nbeg; n0 < nend; n0 += 32) {
        __builtin_prefetch(qrow + n0 + 128, 0, 0);
        v16h a, bm;
#pragma unroll
        for (int e = 0; e < 16; ++e) {
            int ka = n0 + a_ksrc(e, half);          // A: contiguous 16 per half
            int kb = n0 + half * 16 + e;            // B: contiguous 16 per half
            a[e]  = mok ? (_Float16)qrow[ka] : (_Float16)0.f;
            bm[e] = mok ? (_Float16)krow[kb] : (_Float16)0.f;
        }
        acc = __builtin_amdgcn_wmma_f32_16x16x32_f16(false, a, false, bm,
                                                     (short)0, acc, false, false);
    }
#pragma unroll
    for (int r = 0; r < 8; ++r) {
        int i = r + half * 8;
        if (i < 8 && m < 8)
            atomicAdd(&attn[(size_t)bh * 64 + i * 8 + m], acc[r]);
    }
}

// scale raw logits by norms * temperature, softmax over j (in-place)
__global__ __launch_bounds__(256) void k_softmax(const float* __restrict__ nrm,
                                                 const float* __restrict__ temp,
                                                 float* __restrict__ attn) {
    int r = blockIdx.x * 256 + threadIdx.x;   // 512 rows
    if (r >= B_ * HEADS_ * D_) return;
    int b = r >> 6, head = (r >> 3) & 7, i = r & 7;
    float* row = attn + ((size_t)(b * HEADS_ + head)) * 64 + i * 8;
    float qn = nrm[b * C_ + head * D_ + i];
    float t  = temp[head];
    float lg[8];
#pragma unroll
    for (int j = 0; j < 8; ++j)
        lg[j] = row[j] * qn * nrm[512 + b * C_ + head * D_ + j] * t;
    float m = lg[0];
#pragma unroll
    for (int j = 1; j < 8; ++j) m = fmaxf(m, lg[j]);
    float e[8], s = 0.f;
#pragma unroll
    for (int j = 0; j < 8; ++j) { e[j] = __expf(lg[j] - m); s += e[j]; }
    float inv = 1.f / s;
#pragma unroll
    for (int j = 0; j < 8; ++j) row[j] = e[j] * inv;
}

// ---------------- out = attn @ v_s, scattered back to original order via vidx
__global__ __launch_bounds__(256) void k_av_scatter(const float* __restrict__ attn,
                                                    const float* __restrict__ v_s,
                                                    const int* __restrict__ vidx,
                                                    float* __restrict__ outn) {
    size_t gid = (size_t)blockIdx.x * 256 + threadIdx.x;   // B*C*N
    int p = (int)(gid & (N_ - 1));
    int row = (int)(gid >> 16);
    int b = row >> 6, c = row & 63;
    int head = c >> 3, i = c & 7;
    const float* arow = attn + ((size_t)(b * HEADS_ + head)) * 64 + i * 8;
    const float* vb   = v_s + ((size_t)b * C_ + head * D_) * N_;
    float acc = 0.f;
#pragma unroll
    for (int j = 0; j < 8; ++j) acc += arow[j] * vb[(size_t)j * N_ + p];
    outn[((size_t)row << 16) + vidx[gid]] = acc;
}

// ----------------------------------------- channel mean/max, tiny FC, gate ca
__global__ __launch_bounds__(256) void k_chstats(const float* __restrict__ outn,
                                                 float* __restrict__ mean,
                                                 float* __restrict__ mx) {
    __shared__ float rs[256], rm[256];
    const int row = blockIdx.x;          // b*C + c (512)
    const float* src = outn + ((size_t)row << 16);
    float s = 0.f, m = -3.4e38f;
    for (int n = threadIdx.x; n < N_; n += 256) {
        float v = src[n];
        s += v; m = fmaxf(m, v);
    }
    rs[threadIdx.x] = s; rm[threadIdx.x] = m;
    __syncthreads();
    for (int off = 128; off; off >>= 1) {
        if (threadIdx.x < off) {
            rs[threadIdx.x] += rs[threadIdx.x + off];
            rm[threadIdx.x] = fmaxf(rm[threadIdx.x], rm[threadIdx.x + off]);
        }
        __syncthreads();
    }
    if (threadIdx.x == 0) { mean[row] = rs[0] * (1.f / N_); mx[row] = rm[0]; }
}

__global__ __launch_bounds__(64) void k_ca(const float* __restrict__ mean,
                                           const float* __restrict__ mx,
                                           const float* __restrict__ w_fc1,
                                           const float* __restrict__ w_fc2,
                                           float* __restrict__ ca) {
    __shared__ float sm[64], sx[64];
    const int b = blockIdx.x, c = threadIdx.x;
    sm[c] = mean[b * 64 + c];
    sx[c] = mx[b * 64 + c];
    __syncthreads();
    float hm[4], hx[4];
#pragma unroll
    for (int r = 0; r < 4; ++r) {
        float am = 0.f, ax = 0.f;
        for (int k = 0; k < 64; ++k) {
            float w1 = w_fc1[r * 64 + k];
            am += w1 * sm[k];
            ax += w1 * sx[k];
        }
        hm[r] = silu_(am); hx[r] = silu_(ax);
    }
    float om = 0.f, ox = 0.f;
#pragma unroll
    for (int r = 0; r < 4; ++r) {
        om += w_fc2[c * 4 + r] * hm[r];
        ox += w_fc2[c * 4 + r] * hx[r];
    }
    ca[b * 64 + c] = sigm_(om) + sigm_(ox);
}

// -------------------------- oc = outn*ca; spatial stat maps (mean,max,min,sum)
__global__ __launch_bounds__(256) void k_scale_sp(const float* __restrict__ outn,
                                                  const float* __restrict__ ca,
                                                  float* __restrict__ oc,
                                                  float* __restrict__ sp) {
    size_t gid = (size_t)blockIdx.x * 256 + threadIdx.x;   // B*N
    int p = (int)(gid & (N_ - 1));
    int b = (int)(gid >> 16);
    float sm = 0.f, mx = -3.4e38f, mn = 3.4e38f;
    for (int c = 0; c < C_; ++c) {
        float v = outn[((size_t)(b * C_ + c) << 16) + p] * ca[b * C_ + c];
        oc[((size_t)(b * C_ + c) << 16) + p] = v;
        sm += v; mx = fmaxf(mx, v); mn = fminf(mn, v);
    }
    sp[((size_t)(b * 4 + 0) << 16) + p] = sm * (1.f / C_);
    sp[((size_t)(b * 4 + 1) << 16) + p] = mx;
    sp[((size_t)(b * 4 + 2) << 16) + p] = mn;
    sp[((size_t)(b * 4 + 3) << 16) + p] = sm;
}

// ------------------------------- spatial 7x7 conv + silu + 1x1 conv + sigmoid
__global__ __launch_bounds__(256) void k_sa(const float* __restrict__ sp,
                                            const float* __restrict__ w_sp1,
                                            const float* __restrict__ b_sp1,
                                            const float* __restrict__ w_sp2,
                                            const float* __restrict__ b_sp2,
                                            float* __restrict__ sa) {
    size_t gid = (size_t)blockIdx.x * 256 + threadIdx.x;   // B*N
    int p = (int)(gid & (N_ - 1));
    int b = (int)(gid >> 16);
    int h = p >> 8, w = p & 255;
    float acc = 0.f;
    for (int ci = 0; ci < 4; ++ci) {
        const float* chan = sp + ((size_t)(b * 4 + ci) << 16);
        for (int ky = 0; ky < 7; ++ky) {
            int hy = h + ky - 3;
            if ((unsigned)hy >= (unsigned)H_) continue;
            for (int kx = 0; kx < 7; ++kx) {
                int wx = w + kx - 3;
                if ((unsigned)wx >= (unsigned)W_) continue;
                acc += chan[hy * W_ + wx] * w_sp1[(ci * 7 + ky) * 7 + kx];
            }
        }
    }
    float s1 = silu_(acc + b_sp1[0]);
    sa[gid] = sigm_(s1 * w_sp2[0] + b_sp2[0]);
}

// ------------------------- proj GEMM (64x64)x(64xN), epilogue x sa (fused gate)
__global__ __launch_bounds__(256) void k_proj_gemm(const float* __restrict__ oc,
                                                   const float* __restrict__ w_proj,
                                                   const float* __restrict__ sa,
                                                   float* __restrict__ proj) {
    __shared__ _Float16 lAf[16][64];
    __shared__ _Float16 lBt[128][64];
    const int b  = blockIdx.z;
    const int m0 = blockIdx.y * 16;
    const int n0 = blockIdx.x * 128;
    const int tid = threadIdx.x;
    stage_a_frag(w_proj, m0, lAf, tid);
    for (int i = tid; i < 64 * 128; i += 256) {
        int c = i >> 7, j = i & 127;
        lBt[j][c] = (_Float16)oc[((size_t)b * C_ + c) * N_ + n0 + j];
    }
    __syncthreads();
    const int wave = tid >> 5, lane = tid & 31;
    const int nn = lane & 15, half = lane >> 4;
    const int m = lane & 15;
    const int col_l = wave * 16 + nn;
    v8f acc = {};
#pragma unroll
    for (int ks = 0; ks < 2; ++ks) {
        v16h a  = *(const v16h*)&lAf[m][ks * 32 + half * 16];
        v16h bm = *(const v16h*)&lBt[col_l][ks * 32 + half * 16];
        acc = __builtin_amdgcn_wmma_f32_16x16x32_f16(false, a, false, bm,
                                                     (short)0, acc, false, false);
    }
    const int col   = n0 + col_l;
    const int mbase = m0 + half * 8;
    const float g = sa[(size_t)b * N_ + col];
#pragma unroll
    for (int r = 0; r < 8; ++r)
        proj[((size_t)b * C_ + mbase + r) * N_ + col] = acc[r] * g;
}

// --------------------------------------------- final inverse-sort scatters
__global__ __launch_bounds__(256) void k_out_upper(const float* __restrict__ proj,
                                                   float* __restrict__ dout) {
    size_t gid = (size_t)blockIdx.x * 256 + threadIdx.x;   // B*HALF*N
    int p = (int)(gid & (N_ - 1));
    int row = (int)(gid >> 16);
    int b = row >> 5, ch = HALF_ + (row & 31);
    size_t a = ((size_t)(b * C_ + ch) << 16) + p;
    dout[a] = proj[a];
}

__global__ __launch_bounds__(256) void k_unsort_w(const float* __restrict__ proj,
                                                  const int* __restrict__ idx_w,
                                                  float* __restrict__ t1) {
    size_t gid = (size_t)blockIdx.x * 256 + threadIdx.x;   // B*HALF*N
    int p = (int)(gid & (N_ - 1));
    int row = (int)(gid >> 16);                            // b*HALF + ch
    int b = row >> 5, ch = row & 31;
    int h = p >> 8;
    int wdst = idx_w[gid];
    t1[((size_t)row * H_ + h) * W_ + wdst] =
        proj[((size_t)(b * C_ + ch) << 16) + p];
}

__global__ __launch_bounds__(256) void k_unsort_h(const float* __restrict__ t1,
                                                  const int* __restrict__ idx_h,
                                                  float* __restrict__ dout) {
    size_t gid = (size_t)blockIdx.x * 256 + threadIdx.x;   // B*HALF*N
    int p = (int)(gid & (N_ - 1));
    int row = (int)(gid >> 16);
    int b = row >> 5, ch = row & 31;
    int w = p & 255;
    int hdst = idx_h[gid];
    dout[((size_t)(b * C_ + ch) * H_ + hdst) * W_ + w] = t1[((size_t)row << 16) + p];
}

// =============================================================================
extern "C" void kernel_launch(void* const* d_in, const int* in_sizes, int n_in,
                              void* d_out, int out_size, void* d_ws, size_t ws_size,
                              hipStream_t stream) {
    (void)in_sizes; (void)n_in; (void)out_size; (void)ws_size;

    const float* x      = (const float*)d_in[0];
    const float* temp   = (const float*)d_in[1];
    const float* w_qkv  = (const float*)d_in[2];
    const float* w_dw   = (const float*)d_in[3];
    const float* w_fc1  = (const float*)d_in[4];
    const float* w_fc2  = (const float*)d_in[5];
    const float* w_sp1  = (const float*)d_in[6];
    const float* b_sp1  = (const float*)d_in[7];
    const float* w_sp2  = (const float*)d_in[8];
    const float* b_sp2  = (const float*)d_in[9];
    const float* w_proj = (const float*)d_in[10];
    float* dout = (float*)d_out;

    char* wp = (char*)d_ws;
    auto alloc = [&](size_t bytes) -> char* {
        char* p = wp;
        wp += (bytes + 255) & ~(size_t)255;
        return p;
    };
    const size_t BCN = (size_t)B_ * C_ * N_;
    const size_t BHN = (size_t)B_ * HALF_ * N_;
    const size_t BON = (size_t)B_ * OC3_ * N_;

    float* xw    = (float*)alloc(BCN * 4);
    int*   idx_h = (int*)  alloc(BHN * 4);
    int*   idx_w = (int*)  alloc(BHN * 4);
    float* qkv   = (float*)alloc(BON * 4);
    float* qkvd  = (float*)alloc(BON * 4);
    float* vkey  = (float*)alloc(BCN * 4);           // becomes v_s after sort
    int*   vidx  = (int*)  alloc(BCN * 4);
    float* q_s   = (float*)alloc(BCN * 4);
    float* k_s   = (float*)alloc(BCN * 4);
    float* outn  = (float*)alloc(BCN * 4);
    float* oc    = (float*)alloc(BCN * 4);
    float* sp    = (float*)alloc((size_t)B_ * 4 * N_ * 4);
    float* sa    = (float*)alloc((size_t)B_ * N_ * 4);
    float* proj  = (float*)alloc(BCN * 4);
    float* t1    = (float*)alloc(BHN * 4);
    float* nrm   = (float*)alloc(1024 * 4);
    float* attn  = (float*)alloc((size_t)B_ * HEADS_ * 64 * 4);
    float* cmean = (float*)alloc(512 * 4);
    float* cmax  = (float*)alloc(512 * 4);
    float* ca    = (float*)alloc(512 * 4);

    const int blkBCN = (int)(BCN / 256);   // 131072
    const int blkBHN = (int)(BHN / 256);   //  65536
    const int blkBON = (int)(BON / 256);   // 393216
    const int blkBN  = (int)((size_t)B_ * N_ / 256);  // 2048

    // 1) copy x, histogram-sort first half channels along H then W
    k_copy<<<blkBCN, 256, 0, stream>>>(x, xw);
    k_sort_h<<<B_ * HALF_ * W_, 256, 0, stream>>>(xw, idx_h);
    k_sort_w<<<B_ * HALF_ * H_, 256, 0, stream>>>(xw, idx_w);

    // 2) qkv 1x1 conv as WMMA GEMM (TDM-staged B tiles), then depthwise 3x3
    k_qkv_gemm<<<dim3(N_ / 128, OC3_ / 16, B_), 256, 0, stream>>>(xw, w_qkv, qkv);
    k_dwconv<<<blkBON, 256, 0, stream>>>(qkv, w_dw, qkvd);

    // 3) per-channel argsort of v over N (global bitonic, L2-resident passes)
    k_vinit<<<blkBCN, 256, 0, stream>>>(qkvd, vkey, vidx);
    for (int k = 2; k <= N_; k <<= 1)
        for (int j = k >> 1; j > 0; j >>= 1)
            k_bitonic_pass<<<blkBCN, 256, 0, stream>>>(vkey, vidx, k, j);
    k_gather_qk<<<blkBCN, 256, 0, stream>>>(qkvd, vidx, q_s, k_s);

    // 4) attention: norms, split-K WMMA logits + atomics, softmax, AV + scatter
    k_norms<<<1024, 256, 0, stream>>>(q_s, k_s, nrm);
    k_zero<<<16, 256, 0, stream>>>(attn, B_ * HEADS_ * 64);
    k_attn_partial<<<dim3(32, B_ * HEADS_), 32, 0, stream>>>(q_s, k_s, attn);
    k_softmax<<<2, 256, 0, stream>>>(nrm, temp, attn);
    k_av_scatter<<<blkBCN, 256, 0, stream>>>(attn, vkey, vidx, outn);

    // 5) channel attention
    k_chstats<<<512, 256, 0, stream>>>(outn, cmean, cmax);
    k_ca<<<B_, 64, 0, stream>>>(cmean, cmax, w_fc1, w_fc2, ca);

    // 6) spatial attention maps + gate
    k_scale_sp<<<blkBN, 256, 0, stream>>>(outn, ca, oc, sp);
    k_sa<<<blkBN, 256, 0, stream>>>(sp, w_sp1, b_sp1, w_sp2, b_sp2, sa);

    // 7) proj GEMM with fused spatial gate epilogue
    k_proj_gemm<<<dim3(N_ / 128, C_ / 16, B_), 256, 0, stream>>>(oc, w_proj, sa, proj);

    // 8) final inverse histogram sorts (scatter) + upper-half copy
    k_out_upper<<<blkBHN, 256, 0, stream>>>(proj, dout);
    k_unsort_w<<<blkBHN, 256, 0, stream>>>(proj, idx_w, t1);
    k_unsort_h<<<blkBHN, 256, 0, stream>>>(t1, idx_h, dout);
}
